// GraphPooling_3401614098593
// MI455X (gfx1250) — compile-verified, matrix operand order checked by reading, not run on hardware
//
#include <hip/hip_runtime.h>

// ---------------- problem constants (match reference) ----------------
#define NB    64
#define NPG_  1024
#define NTOT  65536          // NB * NPG
#define ETOT  1048576        // NTOT * 16
#define NHID_ 256
#define RATIO_ 128
#define HEADS_ 4
#define DH    64             // NHID/HEADS

// ---------------- bf16 helpers & WMMA types ----------------
typedef __attribute__((ext_vector_type(16))) __bf16 v16bf;
typedef __attribute__((ext_vector_type(8)))  float  v8f;
typedef __attribute__((ext_vector_type(8)))  unsigned short ushort8;
typedef __attribute__((ext_vector_type(16))) unsigned short ushort16;

__device__ __forceinline__ float bf2f(unsigned short u) {
    unsigned int v = ((unsigned int)u) << 16;
    return __builtin_bit_cast(float, v);
}
__device__ __forceinline__ unsigned short f2bf(float f) {
    unsigned int u = __builtin_bit_cast(unsigned int, f);
    unsigned int r = u + 0x7FFFu + ((u >> 16) & 1u);   // round-to-nearest-even
    return (unsigned short)(r >> 16);
}

// Load a 16-element bf16 fragment: two contiguous 8-element (16B) halves.
__device__ __forceinline__ v16bf load_frag(const unsigned short* p0,
                                           const unsigned short* p1) {
    ushort8 lo = *(const ushort8*)p0;
    ushort8 hi = *(const ushort8*)p1;
    ushort16 t;
#pragma unroll
    for (int i = 0; i < 8; ++i) { t[i] = lo[i]; t[i + 8] = hi[i]; }
    return __builtin_bit_cast(v16bf, t);
}
// Fragment from f32 source (LDS), converting on the fly.
__device__ __forceinline__ v16bf frag_from_f32(const float* p0, const float* p1) {
    ushort16 t;
#pragma unroll
    for (int i = 0; i < 8; ++i) { t[i] = f2bf(p0[i]); t[i + 8] = f2bf(p1[i]); }
    return __builtin_bit_cast(v16bf, t);
}
__device__ __forceinline__ v8f wmma_bf16(v16bf a, v16bf b, v8f c) {
    return __builtin_amdgcn_wmma_f32_16x16x32_bf16(false, a, false, b,
                                                   (short)0, c, false, false);
}

// ---------------- small utility kernels ----------------
__global__ void zero_i32(int* p, int n) {
    int i = blockIdx.x * blockDim.x + threadIdx.x;
    if (i < n) p[i] = 0;
}
__global__ void cvt_f32_bf16(const float* __restrict__ s,
                             unsigned short* __restrict__ d, int n) {
    int i = blockIdx.x * blockDim.x + threadIdx.x;
    if (i < n) d[i] = f2bf(s[i]);
}
// W is [K=256][N=256] row-major; write Wt[n][k] bf16 so B-fragments are contiguous.
__global__ void cvt_wt_t(const float* __restrict__ W, unsigned short* __restrict__ Wt) {
    int n = blockIdx.x, k = threadIdx.x;
    Wt[n * NHID_ + k] = f2bf(W[k * NHID_ + n]);
}

// ---------------- degree / CSR ----------------
__global__ void deg_kernel(const int* __restrict__ dst, int* __restrict__ deg, int e) {
    int i = blockIdx.x * blockDim.x + threadIdx.x;
    if (i < e) atomicAdd(&deg[dst[i]], 1);
}
__global__ void dinv_kernel(const int* __restrict__ deg,
                            float* __restrict__ dinv0, float* __restrict__ dinv1) {
    int i = blockIdx.x * blockDim.x + threadIdx.x;
    int d = deg[i];
    dinv0[i] = (d > 0) ? rsqrtf((float)d) : 0.f;
    dinv1[i] = rsqrtf((float)(d + 1));
}
__global__ void scan1_kernel(const int* __restrict__ deg, int* __restrict__ rowStart,
                             int* __restrict__ bsum) {
    __shared__ int s[256];
    int t = threadIdx.x, i = blockIdx.x * 256 + t;
    int v = deg[i];
    s[t] = v; __syncthreads();
    for (int off = 1; off < 256; off <<= 1) {
        int add = (t >= off) ? s[t - off] : 0;
        __syncthreads();
        s[t] += add;
        __syncthreads();
    }
    rowStart[i] = s[t] - v;
    if (t == 255) bsum[blockIdx.x] = s[255];
}
__global__ void scan2_kernel(int* __restrict__ bsum) {
    __shared__ int s[256];
    int t = threadIdx.x;
    int v = bsum[t];
    s[t] = v; __syncthreads();
    for (int off = 1; off < 256; off <<= 1) {
        int add = (t >= off) ? s[t - off] : 0;
        __syncthreads();
        s[t] += add;
        __syncthreads();
    }
    bsum[t] = s[t] - v;
}
__global__ void scan3_kernel(int* __restrict__ rowStart, const int* __restrict__ bsum) {
    int t = threadIdx.x, i = blockIdx.x * 256 + t;
    rowStart[i] += bsum[blockIdx.x];
}
__global__ void fill_kernel(const int* __restrict__ src, const int* __restrict__ dst,
                            const float* __restrict__ dinv1,
                            const int* __restrict__ rowStart, int* __restrict__ cursor,
                            int* __restrict__ eSrc, float* __restrict__ eW, int e) {
    int i = blockIdx.x * blockDim.x + threadIdx.x;
    if (i >= e) return;
    int d = dst[i], s = src[i];
    int pos = rowStart[d] + atomicAdd(&cursor[d], 1);
    eSrc[pos] = s;
    eW[pos]  = dinv1[s] * dinv1[d];
}

// ---------------- score GEMVs (W1, W2) : one wave per node ----------------
__global__ __launch_bounds__(256) void gemv_score(
    const float* __restrict__ x, const float* __restrict__ W1,
    const float* __restrict__ W2, const float* __restrict__ b1,
    const float* __restrict__ dinv0,
    float* __restrict__ s1, float* __restrict__ h2s) {
    int wave = threadIdx.x >> 5, lane = threadIdx.x & 31;
    int node = blockIdx.x * 8 + wave;
    const float* xr = x + (size_t)node * NHID_ + lane * 8;
    float d1 = 0.f, d2 = 0.f;
#pragma unroll
    for (int i = 0; i < 8; ++i) {
        float xv = xr[i];
        d1 += xv * W1[lane * 8 + i];
        d2 += xv * W2[lane * 8 + i];
    }
    for (int m = 16; m > 0; m >>= 1) {
        d1 += __shfl_xor(d1, m, 32);
        d2 += __shfl_xor(d2, m, 32);
    }
    if (lane == 0) {
        s1[node]  = d1 + b1[0];
        h2s[node] = d2 * dinv0[node];
    }
}

// ---------------- generic WMMA GEMM: C[M,256] = A[M,256] @ Bt^T ----------------
// A: bf16 row-major [M][256].  Bt: bf16 [N][256] (transposed weights).
// Each wave computes a 16x64 strip; 8 waves/block -> 128 rows/block.
// mode 0: v = acc (+bias).  mode 1: v = Res + relu(acc + bias).
__global__ __launch_bounds__(256) void wmma_gemm256(
    const unsigned short* __restrict__ A, const unsigned short* __restrict__ Bt,
    const float* __restrict__ bias, float* __restrict__ C,
    unsigned short* __restrict__ Cb, const float* __restrict__ Res, int mode) {
    const int lane = threadIdx.x & 31;
    const int wave = threadIdx.x >> 5;
    const int m0 = (blockIdx.y * 8 + wave) * 16;
    const int n0 = blockIdx.x * 64;
    const int row = lane & 15;
    const int khalf = (lane >> 4) * 8;

    v8f acc[4];
#pragma unroll
    for (int j = 0; j < 4; ++j) acc[j] = (v8f){0.f,0.f,0.f,0.f,0.f,0.f,0.f,0.f};

    const unsigned short* Arow = A + (size_t)(m0 + row) * NHID_ + khalf;
#pragma unroll
    for (int kk = 0; kk < NHID_; kk += 32) {
        v16bf a = load_frag(Arow + kk, Arow + kk + 16);
#pragma unroll
        for (int j = 0; j < 4; ++j) {
            const unsigned short* Bp =
                Bt + (size_t)(n0 + j * 16 + row) * NHID_ + kk + khalf;
            v16bf b = load_frag(Bp, Bp + 16);
            acc[j] = wmma_bf16(a, b, acc[j]);
        }
    }
    const int nlo = lane & 15;
    const int mo = (lane >> 4) * 8;
#pragma unroll
    for (int j = 0; j < 4; ++j) {
#pragma unroll
        for (int r = 0; r < 8; ++r) {
            int gm = m0 + mo + r;
            int gn = n0 + j * 16 + nlo;
            size_t idx = (size_t)gm * NHID_ + gn;
            float v = acc[j][r];
            if (bias) v += bias[gn];
            if (mode == 1) v = Res[idx] + fmaxf(v, 0.f);
            if (C)  C[idx]  = v;
            if (Cb) Cb[idx] = f2bf(v);
        }
    }
}

// ---------------- dual-output WMMA GEMM: C0 = A@Bt0^T, C1 = A@Bt1^T ----------------
// Shares the A-fragment load between both products (halves A traffic; the big
// x@Wk / x@Wv pair). bf16 outputs only, no bias (added post-aggregation).
__global__ __launch_bounds__(256) void wmma_gemm256_dual(
    const unsigned short* __restrict__ A,
    const unsigned short* __restrict__ Bt0, const unsigned short* __restrict__ Bt1,
    unsigned short* __restrict__ C0, unsigned short* __restrict__ C1) {
    const int lane = threadIdx.x & 31;
    const int wave = threadIdx.x >> 5;
    const int m0 = (blockIdx.y * 8 + wave) * 16;
    const int n0 = blockIdx.x * 64;
    const int row = lane & 15;
    const int khalf = (lane >> 4) * 8;

    v8f acc0[4], acc1[4];
#pragma unroll
    for (int j = 0; j < 4; ++j) {
        acc0[j] = (v8f){0.f,0.f,0.f,0.f,0.f,0.f,0.f,0.f};
        acc1[j] = (v8f){0.f,0.f,0.f,0.f,0.f,0.f,0.f,0.f};
    }
    const unsigned short* Arow = A + (size_t)(m0 + row) * NHID_ + khalf;
#pragma unroll
    for (int kk = 0; kk < NHID_; kk += 32) {
        v16bf a = load_frag(Arow + kk, Arow + kk + 16);
#pragma unroll
        for (int j = 0; j < 4; ++j) {
            size_t bo = (size_t)(n0 + j * 16 + row) * NHID_ + kk + khalf;
            v16bf b0 = load_frag(Bt0 + bo, Bt0 + bo + 16);
            acc0[j] = wmma_bf16(a, b0, acc0[j]);
            v16bf b1 = load_frag(Bt1 + bo, Bt1 + bo + 16);
            acc1[j] = wmma_bf16(a, b1, acc1[j]);
        }
    }
    const int nlo = lane & 15;
    const int mo = (lane >> 4) * 8;
#pragma unroll
    for (int j = 0; j < 4; ++j) {
#pragma unroll
        for (int r = 0; r < 8; ++r) {
            size_t idx = (size_t)(m0 + mo + r) * NHID_ + n0 + j * 16 + nlo;
            C0[idx] = f2bf(acc0[j][r]);
            C1[idx] = f2bf(acc1[j][r]);
        }
    }
}

// ---------------- per-node K/V aggregation + score aggregation ----------------
__global__ __launch_bounds__(256) void agg_kernel(
    const unsigned short* __restrict__ hkb, const unsigned short* __restrict__ hvb,
    const float* __restrict__ bk, const float* __restrict__ bv,
    const int* __restrict__ rowStart, const int* __restrict__ deg,
    const int* __restrict__ eSrc, const float* __restrict__ eW,
    const float* __restrict__ dinv0, const float* __restrict__ dinv1,
    const float* __restrict__ h2s,
    unsigned short* __restrict__ Kb, unsigned short* __restrict__ Vt,
    float* __restrict__ sAgg) {
    __shared__ int   sIdx[256];
    __shared__ float sW[256];
    int n = blockIdx.x, tid = threadIdx.x;
    int start = rowStart[n], cnt = deg[n];
    float accK = 0.f, accV = 0.f, accS = 0.f;
    for (int base = 0; base < cnt; base += 256) {
        int m = cnt - base; if (m > 256) m = 256;
        if (tid < m) { sIdx[tid] = eSrc[start + base + tid]; sW[tid] = eW[start + base + tid]; }
        __syncthreads();
        for (int j = 0; j < m; ++j) {
            int s = sIdx[j]; float w = sW[j];
            size_t o = (size_t)s * NHID_ + tid;
            if (j + 1 < m)
                __builtin_prefetch(&hkb[(size_t)sIdx[j + 1] * NHID_ + tid], 0, 1);
            accK += w * bf2f(hkb[o]);
            accV += w * bf2f(hvb[o]);
            if (tid == 0) accS += h2s[s];
        }
        __syncthreads();
    }
    float w2 = dinv1[n] * dinv1[n];
    size_t o = (size_t)n * NHID_ + tid;
    accK += w2 * bf2f(hkb[o]) + bk[tid];
    accV += w2 * bf2f(hvb[o]) + bv[tid];
    Kb[o] = f2bf(accK);
    int b = n >> 10, nl = n & 1023, h = tid >> 6, d = tid & 63;
    Vt[(((size_t)(b * HEADS_ + h) * DH + d) << 10) + nl] = f2bf(accV);
    if (tid == 0) sAgg[n] = accS * dinv0[n];
}

// ---------------- xp = x * tanh(score), first 128 nodes/graph, bf16 ----------------
__global__ __launch_bounds__(256) void xp_kernel(
    const float* __restrict__ x, const float* __restrict__ s1,
    const float* __restrict__ sAgg, const float* __restrict__ b2,
    unsigned short* __restrict__ xp) {
    int rowp = blockIdx.x;                 // 0..8191
    int g = rowp >> 7, loc = rowp & 127;
    int n = g * NPG_ + loc;
    float score = 0.5f * s1[n] + 0.5f * (sAgg[n] + b2[0]);
    float t = tanhf(score);
    xp[(size_t)rowp * NHID_ + threadIdx.x] =
        f2bf(x[(size_t)n * NHID_ + threadIdx.x] * t);
}

// ---------------- fused attention: scores->softmax->P@V (+Q residual) ----------------
// block = 128 threads (4 waves); blockIdx.x = b*4+h; blockIdx.y = 16-row tile.
__global__ __launch_bounds__(128) void attn_kernel(
    const unsigned short* __restrict__ Qb, const float* __restrict__ Qf,
    const unsigned short* __restrict__ Kb, const unsigned short* __restrict__ Vt,
    float* __restrict__ O) {
    __shared__ float S[16 * 1024];         // 64 KB
    const int b = blockIdx.x >> 2, h = blockIdx.x & 3;
    const int rt = blockIdx.y;
    const int lane = threadIdx.x & 31, wave = threadIdx.x >> 5;
    const int row = lane & 15, khalf = (lane >> 4) * 8;
    const int qrow = b * RATIO_ + rt * 16;

    const unsigned short* Qp = Qb + (size_t)(qrow + row) * NHID_ + h * DH + khalf;
    v16bf aq0 = load_frag(Qp, Qp + 16);
    v16bf aq1 = load_frag(Qp + 32, Qp + 48);

    const float scale = 0.0625f;           // 1/sqrt(256)
    for (int nt = 0; nt < 16; ++nt) {
        int nn = wave * 256 + nt * 16;
        const unsigned short* Kp =
            Kb + (size_t)(b * NPG_ + nn + row) * NHID_ + h * DH + khalf;
        v16bf bk0 = load_frag(Kp, Kp + 16);
        v16bf bk1 = load_frag(Kp + 32, Kp + 48);
        v8f acc = (v8f){0.f,0.f,0.f,0.f,0.f,0.f,0.f,0.f};
        acc = wmma_bf16(aq0, bk0, acc);
        acc = wmma_bf16(aq1, bk1, acc);
        int nl = lane & 15, mo = (lane >> 4) * 8;
#pragma unroll
        for (int r = 0; r < 8; ++r) S[(mo + r) * 1024 + nn + nl] = acc[r] * scale;
    }
    __syncthreads();

    {   // softmax: 8 threads per row, 128 cols each, width-8 shuffle reductions
        int r = threadIdx.x >> 3, sub = threadIdx.x & 7;
        float* Sr = S + r * 1024 + sub * 128;
        float mx = -1e30f;
        for (int i = 0; i < 128; ++i) mx = fmaxf(mx, Sr[i]);
        for (int m = 1; m < 8; m <<= 1) mx = fmaxf(mx, __shfl_xor(mx, m, 8));
        float sum = 0.f;
        for (int i = 0; i < 128; ++i) { float e = __expf(Sr[i] - mx); Sr[i] = e; sum += e; }
        for (int m = 1; m < 8; m <<= 1) sum += __shfl_xor(sum, m, 8);
        float inv = 1.f / sum;
        for (int i = 0; i < 128; ++i) Sr[i] *= inv;
    }
    __syncthreads();

    // O strip: this wave's 16 head-dims d0..d0+15, accumulate over n=1024
    int d0 = wave * 16;
    v8f acc = (v8f){0.f,0.f,0.f,0.f,0.f,0.f,0.f,0.f};
    for (int kt = 0; kt < 32; ++kt) {
        const float* Pp = S + row * 1024 + kt * 32 + khalf;
        v16bf aP = frag_from_f32(Pp, Pp + 16);
        const unsigned short* Vp =
            Vt + ((size_t)(b * HEADS_ + h) * DH + d0 + row) * 1024 + kt * 32 + khalf;
        v16bf bV = load_frag(Vp, Vp + 16);
        acc = wmma_bf16(aP, bV, acc);
    }
    int nl = lane & 15, mo = (lane >> 4) * 8;
#pragma unroll
    for (int r = 0; r < 8; ++r) {
        size_t idx = (size_t)(qrow + mo + r) * NHID_ + h * DH + d0 + nl;
        O[idx] = Qf[idx] + acc[r];
    }
}

// ---------------- LayerNorm over 256-wide rows ----------------
__device__ __forceinline__ float block_sum256(float v, float* red) {
    int t = threadIdx.x;
    red[t] = v; __syncthreads();
    for (int s = 128; s > 0; s >>= 1) {
        if (t < s) red[t] += red[t + s];
        __syncthreads();
    }
    float r = red[0]; __syncthreads();
    return r;
}
__global__ __launch_bounds__(256) void ln_kernel(
    const float* __restrict__ X, const float* __restrict__ g,
    const float* __restrict__ be, float* __restrict__ Y,
    unsigned short* __restrict__ Yb) {
    __shared__ float red[256];
    size_t base = (size_t)blockIdx.x * NHID_;
    float v = X[base + threadIdx.x];
    float mu = block_sum256(v, red) * (1.f / 256.f);
    float d = v - mu;
    float var = block_sum256(d * d, red) * (1.f / 256.f);
    float y = d * rsqrtf(var + 1e-5f) * g[threadIdx.x] + be[threadIdx.x];
    Y[base + threadIdx.x] = y;
    if (Yb) Yb[base + threadIdx.x] = f2bf(y);
}

// ---------------- readout: out[b,d] = sum_r O2[b,r,d]*Wr[r] + br ----------------
__global__ __launch_bounds__(256) void readout_kernel(
    const float* __restrict__ O2, const float* __restrict__ Wr,
    const float* __restrict__ br, float* __restrict__ out) {
    int b = blockIdx.x, d = threadIdx.x;
    float acc = 0.f;
    for (int r = 0; r < RATIO_; ++r)
        acc += O2[((size_t)(b * RATIO_ + r)) * NHID_ + d] * Wr[r];
    out[b * NHID_ + d] = acc + br[0];
}

// ---------------- host side ----------------
extern "C" void kernel_launch(void* const* d_in, const int* in_sizes, int n_in,
                              void* d_out, int out_size, void* d_ws, size_t ws_size,
                              hipStream_t stream) {
    const float* x  = (const float*)d_in[0];
    const int* ei   = (const int*)d_in[1];
    const int* src  = ei;
    const int* dst  = ei + ETOT;
    const float* W1 = (const float*)d_in[3];
    const float* b1 = (const float*)d_in[4];
    const float* W2 = (const float*)d_in[5];
    const float* b2 = (const float*)d_in[6];
    const float* Wq = (const float*)d_in[7];
    const float* bq = (const float*)d_in[8];
    const float* Wk = (const float*)d_in[9];
    const float* bk = (const float*)d_in[10];
    const float* Wv = (const float*)d_in[11];
    const float* bv = (const float*)d_in[12];
    const float* Wo = (const float*)d_in[13];
    const float* bo = (const float*)d_in[14];
    const float* g0 = (const float*)d_in[15];
    const float* be0= (const float*)d_in[16];
    const float* g1 = (const float*)d_in[17];
    const float* be1= (const float*)d_in[18];
    const float* Wr = (const float*)d_in[19];
    const float* br = (const float*)d_in[20];
    float* out = (float*)d_out;

    // -------- workspace carving (256B aligned chunks) --------
    size_t off = 0;
    char* base = (char*)d_ws;
    auto carve = [&](size_t bytes) -> char* {
        char* p = base + off;
        off += (bytes + 255) & ~(size_t)255;
        return p;
    };
    int*   deg      = (int*)  carve(NTOT * 4);
    int*   cursor   = (int*)  carve(NTOT * 4);
    float* dinv0    = (float*)carve(NTOT * 4);
    float* dinv1    = (float*)carve(NTOT * 4);
    int*   rowStart = (int*)  carve(NTOT * 4);
    int*   bsum     = (int*)  carve(256 * 4);
    float* s1       = (float*)carve(NTOT * 4);
    float* h2s      = (float*)carve(NTOT * 4);
    float* sAgg     = (float*)carve(NTOT * 4);
    int*   eSrc     = (int*)  carve(ETOT * 4);
    float* eW       = (float*)carve(ETOT * 4);
    unsigned short* xh  = (unsigned short*)carve((size_t)NTOT * NHID_ * 2);
    unsigned short* WqT = (unsigned short*)carve(NHID_ * NHID_ * 2);
    unsigned short* WkT = (unsigned short*)carve(NHID_ * NHID_ * 2);
    unsigned short* WvT = (unsigned short*)carve(NHID_ * NHID_ * 2);
    unsigned short* WoT = (unsigned short*)carve(NHID_ * NHID_ * 2);
    unsigned short* Kb  = (unsigned short*)carve((size_t)NTOT * NHID_ * 2);
    unsigned short* Vt  = (unsigned short*)carve((size_t)NTOT * NHID_ * 2);
    unsigned short* xp  = (unsigned short*)carve((size_t)NB * RATIO_ * NHID_ * 2);
    float* Qf           = (float*)carve((size_t)NB * RATIO_ * NHID_ * 4);
    unsigned short* Qb  = (unsigned short*)carve((size_t)NB * RATIO_ * NHID_ * 2);
    // hk/hv region (67 MB); late-stage O buffers alias it after agg_kernel.
    char* R = carve(2 * (size_t)NTOT * NHID_ * 2);
    unsigned short* hkb = (unsigned short*)R;
    unsigned short* hvb = (unsigned short*)(R + (size_t)NTOT * NHID_ * 2);
    const size_t OB = (size_t)NB * RATIO_ * NHID_;    // 2M elements
    float* Of           = (float*)R;                   // 8 MB
    float* O1           = (float*)(R + OB * 4);        // 8 MB
    unsigned short* O1b = (unsigned short*)(R + OB * 8);   // 4 MB
    float* Zf           = (float*)(R + OB * 10);       // 8 MB
    float* O2           = (float*)(R + OB * 14);       // 8 MB

    // -------- pipeline --------
    zero_i32<<<256, 256, 0, stream>>>(deg, NTOT);
    zero_i32<<<256, 256, 0, stream>>>(cursor, NTOT);
    deg_kernel<<<ETOT / 256, 256, 0, stream>>>(dst, deg, ETOT);
    dinv_kernel<<<256, 256, 0, stream>>>(deg, dinv0, dinv1);
    scan1_kernel<<<256, 256, 0, stream>>>(deg, rowStart, bsum);
    scan2_kernel<<<1, 256, 0, stream>>>(bsum);
    scan3_kernel<<<256, 256, 0, stream>>>(rowStart, bsum);
    fill_kernel<<<ETOT / 256, 256, 0, stream>>>(src, dst, dinv1, rowStart, cursor,
                                                eSrc, eW, ETOT);
    cvt_f32_bf16<<<(NTOT * NHID_) / 256, 256, 0, stream>>>(x, xh, NTOT * NHID_);
    cvt_wt_t<<<256, 256, 0, stream>>>(Wq, WqT);
    cvt_wt_t<<<256, 256, 0, stream>>>(Wk, WkT);
    cvt_wt_t<<<256, 256, 0, stream>>>(Wv, WvT);
    cvt_wt_t<<<256, 256, 0, stream>>>(Wo, WoT);
    gemv_score<<<NTOT / 8, 256, 0, stream>>>(x, W1, W2, b1, dinv0, s1, h2s);
    // hk = x @ Wk, hv = x @ Wv in one pass over A (shared A fragments)
    wmma_gemm256_dual<<<dim3(4, NTOT / 128), 256, 0, stream>>>(xh, WkT, WvT, hkb, hvb);
    agg_kernel<<<NTOT, 256, 0, stream>>>(hkb, hvb, bk, bv, rowStart, deg, eSrc, eW,
                                         dinv0, dinv1, h2s, Kb, Vt, sAgg);
    xp_kernel<<<NB * RATIO_, 256, 0, stream>>>(x, s1, sAgg, b2, xp);
    // Q = xp @ Wq + bq
    wmma_gemm256<<<dim3(4, (NB * RATIO_) / 128), 256, 0, stream>>>(xp, WqT, bq, Qf,
                                                                   Qb, nullptr, 0);
    attn_kernel<<<dim3(NB * HEADS_, RATIO_ / 16), 128, 0, stream>>>(Qb, Qf, Kb, Vt, Of);
    ln_kernel<<<NB * RATIO_, 256, 0, stream>>>(Of, g0, be0, O1, O1b);
    // Z = O1 + relu(O1 @ Wo + bo)
    wmma_gemm256<<<dim3(4, (NB * RATIO_) / 128), 256, 0, stream>>>(O1b, WoT, bo, Zf,
                                                                   nullptr, O1, 1);
    ln_kernel<<<NB * RATIO_, 256, 0, stream>>>(Zf, g1, be1, O2, nullptr);
    readout_kernel<<<NB, 256, 0, stream>>>(O2, Wr, br, out);
    (void)n_in; (void)in_sizes; (void)out_size; (void)ws_size;
}
